// BaseDeepBSDE_48507360641486
// MI455X (gfx1250) — matrix operand order repaired and checked by reading, not compiled, not run on hardware
//
#include <hip/hip_runtime.h>

typedef __attribute__((ext_vector_type(16))) _Float16 v16h;
typedef __attribute__((ext_vector_type(8)))  _Float16 v8h;
typedef __attribute__((ext_vector_type(8)))  float    v8f;

namespace {
constexpr int   kBatch         = 131072;
constexpr int   kSteps         = 50;
constexpr float kDt            = 0.02f;
constexpr float kSqrtDt        = 0.14142135623730951f;
constexpr int   kWavesPerBlock = 4;                          // 128 threads
constexpr int   kTiles         = kBatch / 16;                // 8192 waves
constexpr int   kBlocks        = kTiles / kWavesPerBlock;    // 2048 blocks
}

__device__ __forceinline__ v8f wmma_f16f32(v16h a, v16h b, v8f c) {
  // D = A(16x32 f16) * B(32x16 f16) + C(16x16 f32)
  return __builtin_amdgcn_wmma_f32_16x16x32_f16(false, a, false, b, (short)0, c,
                                                false, false);
}

// B fragment from LDS weights stored transposed: Wt[N][rowlen] (f16).
// Lane n<16 holds (K = chunk*32 + 0..15, col n); lanes 16..31 hold K+16..31.
__device__ __forceinline__ v16h load_bfrag(const _Float16* Wt, int rowlen,
                                           int tile, int chunk, int lane) {
  const int col = tile * 16 + (lane & 15);
  const int kb  = chunk * 32 + ((lane & 16) ? 16 : 0);
  const _Float16* p = Wt + col * rowlen + kb;
  v8h lo = *(const v8h*)(p);
  v8h hi = *(const v8h*)(p + 8);
  v16h r;
#pragma unroll
  for (int i = 0; i < 8; ++i) { r[i] = lo[i]; r[i + 8] = hi[i]; }
  return r;
}

// A fragment from wave-private LDS activation tile H[16][64] (f16).
// Lane m<16: K = chunk*32 + 0..15; lanes 16..31: K = chunk*32 + 16..31.
__device__ __forceinline__ v16h load_afrag(const _Float16* H, int chunk, int lane) {
  const int m  = lane & 15;
  const int kb = chunk * 32 + ((lane & 16) ? 16 : 0);
  const _Float16* p = H + m * 64 + kb;
  v8h lo = *(const v8h*)(p);
  v8h hi = *(const v8h*)(p + 8);
  v16h r;
#pragma unroll
  for (int i = 0; i < 8; ++i) { r[i] = lo[i]; r[i + 8] = hi[i]; }
  return r;
}

// 3-layer MLP on one 16-row tile: x(16x4+bias-row, padded to K=32) -> 64 -> 64
// -> 16(pad). Layer-1 bias is folded into W1t's K=4 row (a1[4] == 1).
// Layers 2/3 add per-lane register biases in the relu pass. Returns D (f32).
__device__ __forceinline__ v8f mlp3(v16h a1,
                                    const _Float16* W1t,
                                    const _Float16* W2t, const float* b2,
                                    const _Float16* W3t, float b3,
                                    _Float16* h1, _Float16* h2, int lane) {
  const int nloc  = lane & 15;
  const int mrow0 = (lane & 16) ? 8 : 0;

  // layer 1: one K-chunk (zero padded beyond K=5; K=4 is the bias row)
#pragma unroll
  for (int t = 0; t < 4; ++t) {
    v8f c = {};
    c = wmma_f16f32(a1, load_bfrag(W1t, 32, t, 0, lane), c);
#pragma unroll
    for (int r = 0; r < 8; ++r) {
      float v = c[r];
      v = v > 0.f ? v : 0.f;
      h1[(mrow0 + r) * 64 + t * 16 + nloc] = (_Float16)v;
    }
  }
  // layer 2: K=64 -> two chunks
  v16h a20 = load_afrag(h1, 0, lane);
  v16h a21 = load_afrag(h1, 1, lane);
#pragma unroll
  for (int t = 0; t < 4; ++t) {
    v8f c = {};
    c = wmma_f16f32(a20, load_bfrag(W2t, 64, t, 0, lane), c);
    c = wmma_f16f32(a21, load_bfrag(W2t, 64, t, 1, lane), c);
#pragma unroll
    for (int r = 0; r < 8; ++r) {
      float v = c[r] + b2[t];
      v = v > 0.f ? v : 0.f;
      h2[(mrow0 + r) * 64 + t * 16 + nloc] = (_Float16)v;
    }
  }
  // layer 3: N padded to 16 (z uses cols 0..2, q uses col 0)
  v16h a30 = load_afrag(h2, 0, lane);
  v16h a31 = load_afrag(h2, 1, lane);
  v8f c = {};
  c = wmma_f16f32(a30, load_bfrag(W3t, 64, 0, 0, lane), c);
  c = wmma_f16f32(a31, load_bfrag(W3t, 64, 0, 1, lane), c);
#pragma unroll
  for (int r = 0; r < 8; ++r) c[r] += b3;
  return c;
}

__global__ __launch_bounds__(128) void bsde_fused_kernel(
    const float* __restrict__ y0,  const float* __restrict__ /*Y0*/,
    const float* __restrict__ qW1, const float* __restrict__ qb1,
    const float* __restrict__ qW2, const float* __restrict__ qb2,
    const float* __restrict__ qW3, const float* __restrict__ qb3,
    const float* __restrict__ zW1, const float* __restrict__ zb1,
    const float* __restrict__ zW2, const float* __restrict__ zb2,
    const float* __restrict__ zW3, const float* __restrict__ zb3,
    const float* __restrict__ dW,  float* __restrict__ partials) {
  // Transposed/padded f16 weights, shared by the whole block.
  __shared__ alignas(16) _Float16 Z1t[64 * 32], Q1t[64 * 32];
  __shared__ alignas(16) _Float16 Z2t[64 * 64], Q2t[64 * 64];
  __shared__ alignas(16) _Float16 Z3t[16 * 64], Q3t[16 * 64];
  __shared__ float Bz2[64], Bq2[64], Bz3[16], Bq3[16];
  // Wave-private scratch tiles.
  __shared__ alignas(16) _Float16 H1[kWavesPerBlock][16 * 64];
  __shared__ alignas(16) _Float16 H2[kWavesPerBlock][16 * 64];
  __shared__ float YL[kWavesPerBlock][16 * 4];
  __shared__ float wpart[kWavesPerBlock];

  const int tid = threadIdx.x;

  // ---- stage weights: transpose, convert to f16, zero-pad ----
  // Layer 1: K=0..3 weights, K=4 bias row (matched by a1[4] == 1).
  for (int i = tid; i < 64 * 32; i += blockDim.x) {
    const int n = i >> 5, k = i & 31;
    Z1t[i] = (k < 4) ? (_Float16)zW1[k * 64 + n]
           : (k == 4) ? (_Float16)zb1[n] : (_Float16)0.f;
    Q1t[i] = (k < 4) ? (_Float16)qW1[k * 64 + n]
           : (k == 4) ? (_Float16)qb1[n] : (_Float16)0.f;
  }
  for (int i = tid; i < 64 * 64; i += blockDim.x) {
    const int n = i >> 6, k = i & 63;
    Z2t[i] = (_Float16)zW2[k * 64 + n];
    Q2t[i] = (_Float16)qW2[k * 64 + n];
  }
  for (int i = tid; i < 16 * 64; i += blockDim.x) {
    const int n = i >> 6, k = i & 63;
    Z3t[i] = (n < 3)  ? (_Float16)zW3[k * 3 + n] : (_Float16)0.f;
    Q3t[i] = (n == 0) ? (_Float16)qW3[k]         : (_Float16)0.f;
  }
  for (int i = tid; i < 64; i += blockDim.x) { Bz2[i] = zb2[i]; Bq2[i] = qb2[i]; }
  if (tid < 16) {
    Bz3[tid] = (tid < 3) ? zb3[tid] : 0.f;
    Bq3[tid] = (tid == 0) ? qb3[0] : 0.f;
  }
  __syncthreads();

  const int wid   = tid >> 5;
  const int lane  = tid & 31;
  const int tile  = blockIdx.x * kWavesPerBlock + wid;
  const int row0  = tile * 16;
  const int nloc  = lane & 15;
  const int mrow0 = (lane & 16) ? 8 : 0;
  const int jc    = (nloc < 3) ? nloc : 2;   // clamped component index
  const int base  = lane & 16;               // shuffle group base lane

  _Float16* h1 = H1[wid];
  _Float16* h2 = H2[wid];
  float*    yl = YL[wid];

  // ---- hoist per-lane layer-2/3 biases into registers (loop-invariant) ----
  float bz2r[4], bq2r[4];
#pragma unroll
  for (int t = 0; t < 4; ++t) {
    bz2r[t] = Bz2[t * 16 + nloc];
    bq2r[t] = Bq2[t * 16 + nloc];
  }
  const float bz3r = Bz3[nloc];
  const float bq3r = Bq3[nloc];

  // ---- init state (fragment layout: row = mrow0+r, component = nloc) ----
  const float y00 = y0[0], y01 = y0[1], y02 = y0[2];
  const float yini = (nloc == 0) ? y00 : (nloc == 1) ? y01 : (nloc == 2) ? y02 : 0.f;
  float yv[8], Yv[8];
#pragma unroll
  for (int r = 0; r < 8; ++r) { yv[r] = yini; Yv[r] = 0.f; }
  if (nloc < 3) {
#pragma unroll
    for (int r = 0; r < 8; ++r) yl[(mrow0 + r) * 4 + nloc] = yini;
  }

  // ---- fused time loop ----
#pragma unroll 1
  for (int s = 0; s < kSteps; ++s) {
    const float tscal = (float)s * kDt;
    const float* dwp = dW + ((size_t)s * kBatch + row0) * 3;
    if (s + 1 < kSteps)
      __builtin_prefetch(dW + ((size_t)(s + 1) * kBatch + row0) * 3, 0, 1);

    // build A1 = [t, y0, y1, y2, 1(bias), 0...]; lanes >=16 hold zero K-halves
    v16h a1 = {};
    if (lane < 16) {
      a1[0] = (_Float16)tscal;
      a1[1] = (_Float16)yl[lane * 4 + 0];
      a1[2] = (_Float16)yl[lane * 4 + 1];
      a1[3] = (_Float16)yl[lane * 4 + 2];
      a1[4] = (_Float16)1.0f;              // multiplies the bias row of W1t
    }

    v8f dz = mlp3(a1, Z1t, Z2t, bz2r, Z3t, bz3r, h1, h2, lane);
    v8f dq = mlp3(a1, Q1t, Q2t, bq2r, Q3t, bq3r, h1, h2, lane);

    // dW loads (all lanes, clamped component -> EXEC stays full)
    float dws[8];
#pragma unroll
    for (int r = 0; r < 8; ++r)
      dws[r] = dwp[(mrow0 + r) * 3 + jc] * kSqrtDt;

    // SDE update
#pragma unroll
    for (int r = 0; r < 8; ++r) {
      const float q  = __shfl(dq[r], base, 32);                 // broadcast q (col 0)
      const float zd = dz[r] * dws[r];
      const float zsum = __shfl(zd, base, 32) + __shfl(zd, base + 1, 32) +
                         __shfl(zd, base + 2, 32);
      float y = yv[r];
      const float sig = 0.2f + 0.1f * tanhf(y);
      y = y + (q - y) * kDt + sig * dws[r];
      yv[r] = y;
      Yv[r] = Yv[r] - 0.5f * q * q * kDt + zsum;
    }
    // publish y for next step's A1 build
    if (nloc < 3) {
#pragma unroll
      for (int r = 0; r < 8; ++r) yl[(mrow0 + r) * 4 + nloc] = yv[r];
    }
  }

  // ---- terminal cost + per-wave partial of sum((Y - |y|^2)^2) ----
  float val = 0.f;
#pragma unroll
  for (int r = 0; r < 8; ++r) {
    const float sq = yv[r] * yv[r];
    const float term = __shfl(sq, base, 32) + __shfl(sq, base + 1, 32) +
                       __shfl(sq, base + 2, 32);
    const float d = Yv[r] - term;
    val += d * d;
  }
  const float wsum = __shfl(val, 0, 32) + __shfl(val, 16, 32);
  if (lane == 0) wpart[wid] = wsum;
  __syncthreads();
  if (tid == 0)
    partials[blockIdx.x] = wpart[0] + wpart[1] + wpart[2] + wpart[3];
}

__global__ __launch_bounds__(256) void bsde_reduce_kernel(
    const float* __restrict__ partials, float* __restrict__ out) {
  __shared__ float sm[256];
  float s = 0.f;
  for (int i = threadIdx.x; i < kBlocks; i += 256) s += partials[i];
  sm[threadIdx.x] = s;
  __syncthreads();
  for (int off = 128; off > 0; off >>= 1) {
    if (threadIdx.x < off) sm[threadIdx.x] += sm[threadIdx.x + off];
    __syncthreads();
  }
  if (threadIdx.x == 0) out[0] = sm[0] / (float)kBatch;  // + total_res (== 0)
}

extern "C" void kernel_launch(void* const* d_in, const int* in_sizes, int n_in,
                              void* d_out, int out_size, void* d_ws, size_t ws_size,
                              hipStream_t stream) {
  (void)in_sizes; (void)n_in; (void)out_size; (void)ws_size;
  const float* y0  = (const float*)d_in[0];
  const float* Y0  = (const float*)d_in[1];
  const float* qW1 = (const float*)d_in[2];
  const float* qb1 = (const float*)d_in[3];
  const float* qW2 = (const float*)d_in[4];
  const float* qb2 = (const float*)d_in[5];
  const float* qW3 = (const float*)d_in[6];
  const float* qb3 = (const float*)d_in[7];
  const float* zW1 = (const float*)d_in[8];
  const float* zb1 = (const float*)d_in[9];
  const float* zW2 = (const float*)d_in[10];
  const float* zb2 = (const float*)d_in[11];
  const float* zW3 = (const float*)d_in[12];
  const float* zb3 = (const float*)d_in[13];
  const float* dW  = (const float*)d_in[14];

  float* partials = (float*)d_ws;            // kBlocks floats, fully rewritten
  float* out      = (float*)d_out;

  bsde_fused_kernel<<<dim3(kBlocks), dim3(128), 0, stream>>>(
      y0, Y0, qW1, qb1, qW2, qb2, qW3, qb3,
      zW1, zb1, zW2, zb2, zW3, zb3, dW, partials);
  bsde_reduce_kernel<<<dim3(1), dim3(256), 0, stream>>>(partials, out);
}
